// SPVCNN_50457275793897
// MI455X (gfx1250) — compile-verified, matrix operand order checked by reading, not run on hardware
//
#include <hip/hip_runtime.h>
#include <climits>

// ---------------------------------------------------------------------------
// SPVCNN forward for MI455X (gfx1250): all conv/linear layers via
// v_wmma_f32_16x16x32_bf16 implicit GEMM, wave32.
// Each wave computes a 32-voxel x 32-channel tile (2x2 WMMA tiles, 4 f32
// accumulators) so each A/B fragment feeds two WMMAs. Grid dims are template
// constants -> no runtime integer division in the hot loop.
// ---------------------------------------------------------------------------

typedef __attribute__((ext_vector_type(16))) __bf16 v16bf;
typedef __attribute__((ext_vector_type(8)))  __bf16 v8bf;
typedef __attribute__((ext_vector_type(8)))  float  v8f;

#define DEV __device__ __forceinline__

DEV __bf16 f2bf(float f) {
    union { float f; unsigned u; } c; c.f = f;
    unsigned u = c.u;
    unsigned short s = (unsigned short)((u + 0x7FFFu + ((u >> 16) & 1u)) >> 16); // RNE
    __bf16 b; __builtin_memcpy(&b, &s, 2); return b;
}
DEV float bf2f(__bf16 b) {
    unsigned short s; __builtin_memcpy(&s, &b, 2);
    union { unsigned u; float f; } c; c.u = ((unsigned)s) << 16; return c.f;
}

// ------------------------------ prep kernels -------------------------------

__global__ void k_fill_i32(int* p, int n, int v) {
    int i = blockIdx.x * blockDim.x + threadIdx.x;
    if (i < n) p[i] = v;
}

__global__ void k_coord_min(const float* __restrict__ x, int n,
                            const int* __restrict__ voxsz, int* __restrict__ pcmin) {
    float inv = 1.0f / (float)voxsz[0];
    int m0 = INT_MAX, m1 = INT_MAX, m2 = INT_MAX;
    for (int i = blockIdx.x * blockDim.x + threadIdx.x; i < n; i += gridDim.x * blockDim.x) {
        m0 = min(m0, (int)rintf(x[3 * i + 0] * inv));
        m1 = min(m1, (int)rintf(x[3 * i + 1] * inv));
        m2 = min(m2, (int)rintf(x[3 * i + 2] * inv));
    }
    __shared__ int s0[256], s1[256], s2[256];
    s0[threadIdx.x] = m0; s1[threadIdx.x] = m1; s2[threadIdx.x] = m2;
    __syncthreads();
    for (int s = 128; s > 0; s >>= 1) {
        if ((int)threadIdx.x < s) {
            s0[threadIdx.x] = min(s0[threadIdx.x], s0[threadIdx.x + s]);
            s1[threadIdx.x] = min(s1[threadIdx.x], s1[threadIdx.x + s]);
            s2[threadIdx.x] = min(s2[threadIdx.x], s2[threadIdx.x + s]);
        }
        __syncthreads();
    }
    if (threadIdx.x == 0) {
        atomicMin(&pcmin[0], s0[0]); atomicMin(&pcmin[1], s1[0]); atomicMin(&pcmin[2], s2[0]);
    }
}

__global__ void k_scatter_first(const float* __restrict__ x, int n,
                                const int* __restrict__ voxsz,
                                const int* __restrict__ pcmin, int* __restrict__ first) {
    float inv = 1.0f / (float)voxsz[0];
    for (int i = blockIdx.x * blockDim.x + threadIdx.x; i < n; i += gridDim.x * blockDim.x) {
        int d = (int)rintf(x[3 * i + 0] * inv) - pcmin[0];
        int h = (int)rintf(x[3 * i + 1] * inv) - pcmin[1];
        int w = (int)rintf(x[3 * i + 2] * inv) - pcmin[2];
        if ((unsigned)d < 40u && (unsigned)h < 40u && (unsigned)w < 16u)
            atomicMin(&first[(d * 40 + h) * 16 + w], i);
    }
}

__global__ void k_feats(const float* __restrict__ x, const int* __restrict__ first,
                        int npts, float* __restrict__ m1, __bf16* __restrict__ g) {
    int v = blockIdx.x * blockDim.x + threadIdx.x;
    if (v >= 25600) return;
    int f = first[v];
    bool occ = f < npts;
    int idx = min(f, npts - 1);
    float vals[4] = { x[3 * idx + 0], x[3 * idx + 1], x[3 * idx + 2], 1.0f };
    m1[v] = occ ? 1.0f : 0.0f;
    const __bf16 zb = f2bf(0.0f);
#pragma unroll
    for (int c = 0; c < 4; ++c) g[(size_t)v * 32 + c] = occ ? f2bf(vals[c]) : zb;
    for (int c = 4; c < 32; ++c) g[(size_t)v * 32 + c] = zb;
}

__global__ void k_pool_mask(const float* __restrict__ mi, int Do, int Ho, int Wo,
                            float* __restrict__ mo) {
    int v = blockIdx.x * blockDim.x + threadIdx.x;
    if (v >= Do * Ho * Wo) return;
    int w = v % Wo, h = (v / Wo) % Ho, d = v / (Wo * Ho);
    int Hi = 2 * Ho, Wi = 2 * Wo;
    float m = 0.0f;
#pragma unroll
    for (int i = 0; i < 2; ++i)
#pragma unroll
        for (int j = 0; j < 2; ++j)
#pragma unroll
            for (int k = 0; k < 2; ++k)
                m = fmaxf(m, mi[((2 * d + i) * Hi + 2 * h + j) * Wi + 2 * w + k]);
    mo[v] = m;
}

// Pack raw f32 weights (taps, Cin, Cout) into per-lane WMMA-B tiles:
// bp[(((t*Kc+kc)*Nt+nt)*32 + lane)*16 + j], j -> (VGPR j/2, half j&1).
__global__ void k_pack_w(const float* __restrict__ w, int taps, int Cin, int Cout,
                         int Kc, int Nt, __bf16* __restrict__ bp, int total) {
    int e = blockIdx.x * blockDim.x + threadIdx.x;
    if (e >= total) return;
    int j = e & 15, lane = (e >> 4) & 31, r = e >> 9;
    int nt = r % Nt; r /= Nt;
    int kc = r % Kc; int t = r / Kc;
    int v = j >> 1, tt = j & 1;
    int klocal = ((v < 4) ? 0 : 16) + ((lane < 16) ? 0 : 8) + ((v & 3) << 1) + tt;
    int k = kc * 32 + klocal;
    int n = nt * 16 + (lane & 15);
    float val = (k < Cin) ? w[((size_t)t * Cin + k) * Cout + n] : 0.0f;
    bp[e] = f2bf(val);
}

__global__ void k_copy_ch(const __bf16* __restrict__ src, int scs,
                          __bf16* __restrict__ dst, int dcs, int coff,
                          int nvox, int C) {
    int i = blockIdx.x * blockDim.x + threadIdx.x;
    if (i >= nvox * C) return;
    int v = i / C, c = i % C;
    dst[(size_t)v * dcs + coff + c] = src[(size_t)v * scs + c];
}

// ------------------------------ conv kernel --------------------------------
// MODE: 0 = 3x3x3 SAME, 1 = 2x2x2 stride-2 down, 2 = 2x2x2 stride-2 up
// (conv_transpose: tap = flipped output parity), 3 = 1x1 linear.
// One wave per block; wave computes 32 voxels x 32 channels as 2x2 WMMA tiles.
template <int MODE, int D, int H, int W>
__launch_bounds__(32)
__global__ void k_conv_wmma(const __bf16* __restrict__ in, int in_cs,
                            const __bf16* __restrict__ bp, int Kc,
                            const float* __restrict__ bnm, const float* __restrict__ bnv,
                            const float* __restrict__ mask,
                            const __bf16* __restrict__ resid, int res_cs,
                            __bf16* __restrict__ out, int out_cs, int out_coff,
                            float* __restrict__ fout, int relu) {
    constexpr int nvox = D * H * W;
    constexpr int inD = (MODE == 1) ? 2 * D : ((MODE == 2) ? D / 2 : D);
    constexpr int inH = (MODE == 1) ? 2 * H : ((MODE == 2) ? H / 2 : H);
    constexpr int inW = (MODE == 1) ? 2 * W : ((MODE == 2) ? W / 2 : W);
    constexpr int ntaps = (MODE == 0) ? 27 : ((MODE == 3) ? 1 : 8);

    const int lane = threadIdx.x;
    const int mrow = lane & 15;
    const int half = lane >> 4;
    const int Nt = gridDim.y * 2;           // total 16-ch tiles
    const int nt0 = blockIdx.y * 2;
    const __bf16 zb = f2bf(0.0f);

    // Per-wave: two 16-voxel row groups.
    int d_[2], h_[2], w_[2], mytap[2];
    bool vok[2];
#pragma unroll
    for (int r = 0; r < 2; ++r) {
        int vx = blockIdx.x * 32 + r * 16 + mrow;
        vok[r] = vx < nvox;
        int vv = vok[r] ? vx : 0;
        w_[r] = vv % W; int t = vv / W; h_[r] = t % H; d_[r] = t / H;
        mytap[r] = ((1 - (d_[r] & 1)) << 2) | ((1 - (h_[r] & 1)) << 1) | (1 - (w_[r] & 1));
    }

    v8f acc[4] = {};   // [mi*2+ni]
    for (int t = 0; t < ntaps; ++t) {
        const __bf16* prow[2];
        bool ok[2];
#pragma unroll
        for (int r = 0; r < 2; ++r) {
            ok[r] = vok[r];
            int nd = d_[r], nh = h_[r], nw = w_[r];
            if (MODE == 0) {
                nd += t / 9 - 1; nh += (t / 3) % 3 - 1; nw += t % 3 - 1;
                ok[r] = ok[r] && (unsigned)nd < (unsigned)inD &&
                        (unsigned)nh < (unsigned)inH && (unsigned)nw < (unsigned)inW;
            } else if (MODE == 1) {
                nd = 2 * nd + (t >> 2); nh = 2 * nh + ((t >> 1) & 1); nw = 2 * nw + (t & 1);
            } else if (MODE == 2) {
                nd >>= 1; nh >>= 1; nw >>= 1;
                ok[r] = ok[r] && (t == mytap[r]);
            }
            prow[r] = in + (size_t)((nd * inH + nh) * inW + nw) * in_cs;
        }
        const __bf16* bbase = bp + ((size_t)(t * Kc) * Nt) * 512 + lane * 16;
        for (int kc = 0; kc < Kc; ++kc) {
            // A fragments (16-bit 16x32 layout: two 8-elem segments per lane)
            v16bf a[2];
#pragma unroll
            for (int r = 0; r < 2; ++r) {
                if (ok[r]) {
                    const __bf16* pk = prow[r] + kc * 32 + half * 8;
                    v8bf lo = *(const v8bf*)pk;
                    v8bf hi = *(const v8bf*)(pk + 16);
#pragma unroll
                    for (int i = 0; i < 8; ++i) { a[r][i] = lo[i]; a[r][8 + i] = hi[i]; }
                } else {
#pragma unroll
                    for (int i = 0; i < 16; ++i) a[r][i] = zb;
                }
            }
            const __bf16* bk = bbase + (size_t)kc * Nt * 512;
            v16bf b0 = *(const v16bf*)(bk + (size_t)nt0 * 512);
            v16bf b1 = *(const v16bf*)(bk + (size_t)(nt0 + 1) * 512);
            acc[0] = __builtin_amdgcn_wmma_f32_16x16x32_bf16(false, a[0], false, b0,
                                                             (short)0, acc[0], false, false);
            acc[1] = __builtin_amdgcn_wmma_f32_16x16x32_bf16(false, a[0], false, b1,
                                                             (short)0, acc[1], false, false);
            acc[2] = __builtin_amdgcn_wmma_f32_16x16x32_bf16(false, a[1], false, b0,
                                                             (short)0, acc[2], false, false);
            acc[3] = __builtin_amdgcn_wmma_f32_16x16x32_bf16(false, a[1], false, b1,
                                                             (short)0, acc[3], false, false);
        }
    }

    // Epilogue. C/D layout: VGPR r -> M = r + half*8, N = lane&15.
#pragma unroll
    for (int ni = 0; ni < 2; ++ni) {
        const int ch = (nt0 + ni) * 16 + mrow;
        const float bm = bnm[ch];
        const float rs = rsqrtf(bnv[ch] + 1e-5f);
#pragma unroll
        for (int mi = 0; mi < 2; ++mi) {
            const v8f& A = acc[mi * 2 + ni];
#pragma unroll
            for (int r = 0; r < 8; ++r) {
                int v2 = blockIdx.x * 32 + mi * 16 + r + half * 8;
                if (v2 >= nvox) break;
                float val = (A[r] - bm) * rs;
                if (resid) val += bf2f(resid[(size_t)v2 * res_cs + ch]);
                if (relu) val = fmaxf(val, 0.0f);
                val *= mask[v2];
                if (out)  out[(size_t)v2 * out_cs + out_coff + ch] = f2bf(val);
                if (fout) fout[(size_t)ch * nvox + v2] = val;
            }
        }
    }
}

// ------------------------------ host driver --------------------------------

namespace {

struct BN  { const float* m; const float* v; };
struct Res { const float* w1; BN b1; const float* w2; BN b2; const float* wd; BN bd; };
struct Stg { const float* dw; BN db; Res r1, r2; };
struct PW  { const __bf16* p; int Kc; };

// level: 0=(40,40,16) 1=(20,20,8) 2=(10,10,4) 3=(5,5,2)
static const int LD[4] = { 40, 20, 10, 5 };
static const int LH[4] = { 40, 20, 10, 5 };
static const int LW[4] = { 16,  8,  4, 2 };

struct Ctx {
    hipStream_t s;
    char* ws;
    size_t off;
    void* alloc(size_t bytes) {
        void* p = ws + off;
        off = (off + bytes + 255) & ~(size_t)255;
        return p;
    }
    PW pack(const float* w, int taps, int Cin, int Cout) {
        int Kc = (Cin + 31) / 32, Nt = Cout / 16;
        int total = taps * Kc * Nt * 512;
        __bf16* dst = (__bf16*)alloc((size_t)total * sizeof(__bf16));
        k_pack_w<<<(total + 255) / 256, 256, 0, s>>>(w, taps, Cin, Cout, Kc, Nt, dst, total);
        return { dst, Kc };
    }
    void conv(const __bf16* in, int in_cs, PW pw, BN bn, const float* mask,
              const __bf16* resid, int res_cs,
              __bf16* out, int out_cs, int out_coff, float* fout,
              int level, int Cout, int mode, int relu) {
        int nvox = LD[level] * LH[level] * LW[level];
        dim3 g((nvox + 31) / 32, Cout / 32);
        dim3 b(32);
#define KC_ARGS in, in_cs, pw.p, pw.Kc, bn.m, bn.v, mask, resid, res_cs, \
                out, out_cs, out_coff, fout, relu
#define KC_CASE(M, L, DD, HH, WW) \
        case (M * 4 + L): k_conv_wmma<M, DD, HH, WW><<<g, b, 0, s>>>(KC_ARGS); break;
        switch (mode * 4 + level) {
            KC_CASE(0, 0, 40, 40, 16) KC_CASE(0, 1, 20, 20, 8)
            KC_CASE(0, 2, 10, 10, 4)  KC_CASE(0, 3, 5, 5, 2)
            KC_CASE(1, 0, 40, 40, 16) KC_CASE(1, 1, 20, 20, 8)
            KC_CASE(1, 2, 10, 10, 4)  KC_CASE(1, 3, 5, 5, 2)
            KC_CASE(2, 0, 40, 40, 16) KC_CASE(2, 1, 20, 20, 8)
            KC_CASE(2, 2, 10, 10, 4)
            KC_CASE(3, 0, 40, 40, 16) KC_CASE(3, 1, 20, 20, 8)
            KC_CASE(3, 2, 10, 10, 4)  KC_CASE(3, 3, 5, 5, 2)
            default: break;
        }
#undef KC_CASE
#undef KC_ARGS
    }
    void copy(const __bf16* src, int scs, __bf16* dst, int dcs, int coff, int nvox, int C) {
        int tot = nvox * C;
        k_copy_ch<<<(tot + 255) / 256, 256, 0, s>>>(src, scs, dst, dcs, coff, nvox, C);
    }
};

} // namespace

extern "C" void kernel_launch(void* const* d_in, const int* in_sizes, int n_in,
                              void* d_out, int out_size, void* d_ws, size_t ws_size,
                              hipStream_t stream) {
    (void)n_in; (void)out_size; (void)ws_size;
    const float* x = (const float*)d_in[0];
    const int* voxsz = (const int*)d_in[1];
    const int npts = in_sizes[0] / 3;

    // ---- parse params in setup_inputs() flattening order ----
    int ii = 2;
    auto F  = [&]() { return (const float*)d_in[ii++]; };
    auto gB = [&]() { BN b; b.m = F(); b.v = F(); return b; };
    auto gR = [&](bool wd) {
        Res r; r.w1 = F(); r.b1 = gB(); r.w2 = F(); r.b2 = gB();
        r.wd = nullptr; r.bd = { nullptr, nullptr };
        if (wd) { r.wd = F(); r.bd = gB(); }
        return r;
    };
    const float* st_w0 = F(); BN st_b0 = gB();
    const float* st_w1 = F(); BN st_b1 = gB();
    Stg s1; s1.dw = F(); s1.db = gB(); s1.r1 = gR(false); s1.r2 = gR(false);
    Stg s2; s2.dw = F(); s2.db = gB(); s2.r1 = gR(true);  s2.r2 = gR(false);
    Stg s3; s3.dw = F(); s3.db = gB(); s3.r1 = gR(true);  s3.r2 = gR(false);
    Stg u1; u1.dw = F(); u1.db = gB(); u1.r1 = gR(true);  u1.r2 = gR(false);
    Stg u2; u2.dw = F(); u2.db = gB(); u2.r1 = gR(true);  u2.r2 = gR(false);
    Stg u3; u3.dw = F(); u3.db = gB(); u3.r1 = gR(false); u3.r2 = gR(false);

    Ctx C{ stream, (char*)d_ws, 0 };

    const int n0 = 25600, n1 = 3200, n2 = 400, n3 = 50;

    // ---- workspace buffers ----
    int*   pcmin = (int*)C.alloc(4 * sizeof(int));
    int*   first = (int*)C.alloc((size_t)n0 * sizeof(int));
    float* m1 = (float*)C.alloc((size_t)n0 * sizeof(float));
    float* m2 = (float*)C.alloc((size_t)n1 * sizeof(float));
    float* m4 = (float*)C.alloc((size_t)n2 * sizeof(float));
    float* m8 = (float*)C.alloc((size_t)n3 * sizeof(float));
    __bf16* g  = (__bf16*)C.alloc((size_t)n0 * 32 * sizeof(__bf16));
    __bf16* A0 = (__bf16*)C.alloc((size_t)n0 * 256 * sizeof(__bf16));
    __bf16* B0 = (__bf16*)C.alloc((size_t)n0 * 256 * sizeof(__bf16));
    __bf16* C0 = (__bf16*)C.alloc((size_t)n0 * 256 * sizeof(__bf16));
    __bf16* A1 = (__bf16*)C.alloc((size_t)n1 * 128 * sizeof(__bf16));
    __bf16* B1 = (__bf16*)C.alloc((size_t)n1 * 128 * sizeof(__bf16));
    __bf16* C1 = (__bf16*)C.alloc((size_t)n1 * 128 * sizeof(__bf16));
    __bf16* A2 = (__bf16*)C.alloc((size_t)n2 * 192 * sizeof(__bf16));
    __bf16* B2 = (__bf16*)C.alloc((size_t)n2 * 192 * sizeof(__bf16));
    __bf16* C2 = (__bf16*)C.alloc((size_t)n2 * 192 * sizeof(__bf16));
    __bf16* A3 = (__bf16*)C.alloc((size_t)n3 * 128 * sizeof(__bf16));
    __bf16* B3 = (__bf16*)C.alloc((size_t)n3 * 128 * sizeof(__bf16));
    __bf16* C3 = (__bf16*)C.alloc((size_t)n3 * 128 * sizeof(__bf16));

    // ---- pack all weights to WMMA-B layout (bf16) ----
    PW p_st0 = C.pack(st_w0, 27, 4, 32);   // g padded to 32 ch, Kc=1
    PW p_st1 = C.pack(st_w1, 27, 32, 32);
    PW p_s1d = C.pack(s1.dw, 8, 32, 32);
    PW p_s1r1a = C.pack(s1.r1.w1, 27, 32, 32), p_s1r1b = C.pack(s1.r1.w2, 27, 32, 32);
    PW p_s1r2a = C.pack(s1.r2.w1, 27, 32, 32), p_s1r2b = C.pack(s1.r2.w2, 27, 32, 32);
    PW p_s2d = C.pack(s2.dw, 8, 32, 32);
    PW p_s2r1a = C.pack(s2.r1.w1, 27, 32, 64), p_s2r1b = C.pack(s2.r1.w2, 27, 64, 64);
    PW p_s2r1d = C.pack(s2.r1.wd, 1, 32, 64);
    PW p_s2r2a = C.pack(s2.r2.w1, 27, 64, 64), p_s2r2b = C.pack(s2.r2.w2, 27, 64, 64);
    PW p_s3d = C.pack(s3.dw, 8, 64, 64);
    PW p_s3r1a = C.pack(s3.r1.w1, 27, 64, 128), p_s3r1b = C.pack(s3.r1.w2, 27, 128, 128);
    PW p_s3r1d = C.pack(s3.r1.wd, 1, 64, 128);
    PW p_s3r2a = C.pack(s3.r2.w1, 27, 128, 128), p_s3r2b = C.pack(s3.r2.w2, 27, 128, 128);
    PW p_u1d = C.pack(u1.dw, 8, 128, 128);
    PW p_u1r1a = C.pack(u1.r1.w1, 27, 192, 128), p_u1r1b = C.pack(u1.r1.w2, 27, 128, 128);
    PW p_u1r1d = C.pack(u1.r1.wd, 1, 192, 128);
    PW p_u1r2a = C.pack(u1.r2.w1, 27, 128, 128), p_u1r2b = C.pack(u1.r2.w2, 27, 128, 128);
    PW p_u2d = C.pack(u2.dw, 8, 128, 96);
    PW p_u2r1a = C.pack(u2.r1.w1, 27, 128, 96), p_u2r1b = C.pack(u2.r1.w2, 27, 96, 96);
    PW p_u2r1d = C.pack(u2.r1.wd, 1, 128, 96);
    PW p_u2r2a = C.pack(u2.r2.w1, 27, 96, 96), p_u2r2b = C.pack(u2.r2.w2, 27, 96, 96);
    PW p_u3d = C.pack(u3.dw, 8, 96, 256);
    PW p_u3r1a = C.pack(u3.r1.w1, 27, 256, 256), p_u3r1b = C.pack(u3.r1.w2, 27, 256, 256);
    PW p_u3r2a = C.pack(u3.r2.w1, 27, 256, 256), p_u3r2b = C.pack(u3.r2.w2, 27, 256, 256);

    // ---- voxelize ----
    k_fill_i32<<<1, 4, 0, stream>>>(pcmin, 3, INT_MAX);
    k_fill_i32<<<(n0 + 255) / 256, 256, 0, stream>>>(first, n0, npts);
    k_coord_min<<<2048, 256, 0, stream>>>(x, npts, voxsz, pcmin);
    k_scatter_first<<<4096, 256, 0, stream>>>(x, npts, voxsz, pcmin, first);
    k_feats<<<(n0 + 255) / 256, 256, 0, stream>>>(x, first, npts, m1, g);
    k_pool_mask<<<(n1 + 255) / 256, 256, 0, stream>>>(m1, 20, 20, 8, m2);
    k_pool_mask<<<(n2 + 255) / 256, 256, 0, stream>>>(m2, 10, 10, 4, m4);
    k_pool_mask<<<(n3 + 255) / 256, 256, 0, stream>>>(m4, 5, 5, 2, m8);

    float* fout = (float*)d_out;
    const __bf16* NB = nullptr;

    // ---- stem (level 0, 32 ch) ----
    C.conv(g,  32, p_st0, st_b0, m1, NB, 0, A0, 32, 0, nullptr, 0, 32, 0, 1);
    C.conv(A0, 32, p_st1, st_b1, m1, NB, 0, B0, 32, 0, nullptr, 0, 32, 0, 1); // x0=B0

    // ---- s1 (level 1, 32 ch) ----
    C.conv(B0, 32, p_s1d, s1.db, m2, NB, 0, A1, 32, 0, nullptr, 1, 32, 1, 1);
    C.conv(A1, 32, p_s1r1a, s1.r1.b1, m2, NB, 0, B1, 32, 0, nullptr, 1, 32, 0, 1);
    C.conv(B1, 32, p_s1r1b, s1.r1.b2, m2, A1, 32, C1, 32, 0, nullptr, 1, 32, 0, 1);
    C.conv(C1, 32, p_s1r2a, s1.r2.b1, m2, NB, 0, B1, 32, 0, nullptr, 1, 32, 0, 1);
    C.conv(B1, 32, p_s1r2b, s1.r2.b2, m2, C1, 32, A1, 32, 0, nullptr, 1, 32, 0, 1); // x1=A1

    // ---- s2 (level 2, 32->64) ----
    C.conv(A1, 32, p_s2d, s2.db, m4, NB, 0, A2, 32, 0, nullptr, 2, 32, 1, 1);
    C.conv(A2, 32, p_s2r1a, s2.r1.b1, m4, NB, 0, B2, 64, 0, nullptr, 2, 64, 0, 1);
    C.conv(A2, 32, p_s2r1d, s2.r1.bd, m4, NB, 0, C2, 64, 0, nullptr, 2, 64, 3, 0);
    C.conv(B2, 64, p_s2r1b, s2.r1.b2, m4, C2, 64, A2, 64, 0, nullptr, 2, 64, 0, 1);
    C.conv(A2, 64, p_s2r2a, s2.r2.b1, m4, NB, 0, B2, 64, 0, nullptr, 2, 64, 0, 1);
    C.conv(B2, 64, p_s2r2b, s2.r2.b2, m4, A2, 64, C2, 64, 0, nullptr, 2, 64, 0, 1); // x2=C2

    // ---- s3 (level 3, 64->128) ----
    C.conv(C2, 64, p_s3d, s3.db, m8, NB, 0, A3, 64, 0, nullptr, 3, 64, 1, 1);
    C.conv(A3, 64, p_s3r1a, s3.r1.b1, m8, NB, 0, B3, 128, 0, nullptr, 3, 128, 0, 1);
    C.conv(A3, 64, p_s3r1d, s3.r1.bd, m8, NB, 0, C3, 128, 0, nullptr, 3, 128, 3, 0);
    C.conv(B3, 128, p_s3r1b, s3.r1.b2, m8, C3, 128, A3, 128, 0, nullptr, 3, 128, 0, 1);
    C.conv(A3, 128, p_s3r2a, s3.r2.b1, m8, NB, 0, B3, 128, 0, nullptr, 3, 128, 0, 1);
    C.conv(B3, 128, p_s3r2b, s3.r2.b2, m8, A3, 128, C3, 128, 0, nullptr, 3, 128, 0, 1); // x3=C3

    // ---- u1 (level 2, up 128 + concat x2(64) = 192 -> 128) ----
    C.conv(C3, 128, p_u1d, u1.db, m4, NB, 0, A2, 192, 0, nullptr, 2, 128, 2, 1);
    C.copy(C2, 64, A2, 192, 128, n2, 64);
    C.conv(A2, 192, p_u1r1a, u1.r1.b1, m4, NB, 0, B2, 128, 0, nullptr, 2, 128, 0, 1);
    C.conv(A2, 192, p_u1r1d, u1.r1.bd, m4, NB, 0, C2, 128, 0, nullptr, 2, 128, 3, 0);
    C.conv(B2, 128, p_u1r1b, u1.r1.b2, m4, C2, 128, A2, 128, 0, nullptr, 2, 128, 0, 1);
    C.conv(A2, 128, p_u1r2a, u1.r2.b1, m4, NB, 0, B2, 128, 0, nullptr, 2, 128, 0, 1);
    C.conv(B2, 128, p_u1r2b, u1.r2.b2, m4, A2, 128, C2, 128, 0, nullptr, 2, 128, 0, 1); // y=C2

    // ---- u2 (level 1, up 96 + concat x1(32) = 128 -> 96) ----
    C.conv(C2, 128, p_u2d, u2.db, m2, NB, 0, B1, 128, 0, nullptr, 1, 96, 2, 1);
    C.copy(A1, 32, B1, 128, 96, n1, 32);
    C.conv(B1, 128, p_u2r1a, u2.r1.b1, m2, NB, 0, C1, 96, 0, nullptr, 1, 96, 0, 1);
    C.conv(B1, 128, p_u2r1d, u2.r1.bd, m2, NB, 0, A1, 96, 0, nullptr, 1, 96, 3, 0);
    C.conv(C1, 96, p_u2r1b, u2.r1.b2, m2, A1, 96, B1, 96, 0, nullptr, 1, 96, 0, 1);
    C.conv(B1, 96, p_u2r2a, u2.r2.b1, m2, NB, 0, C1, 96, 0, nullptr, 1, 96, 0, 1);
    C.conv(C1, 96, p_u2r2b, u2.r2.b2, m2, B1, 96, A1, 96, 0, nullptr, 1, 96, 0, 1); // y=A1

    // ---- u3 (level 0, up 96->256, two res blocks 256->256) ----
    C.conv(A1, 96, p_u3d, u3.db, m1, NB, 0, A0, 256, 0, nullptr, 0, 256, 2, 1);
    C.conv(A0, 256, p_u3r1a, u3.r1.b1, m1, NB, 0, B0, 256, 0, nullptr, 0, 256, 0, 1);
    C.conv(B0, 256, p_u3r1b, u3.r1.b2, m1, A0, 256, C0, 256, 0, nullptr, 0, 256, 0, 1);
    C.conv(C0, 256, p_u3r2a, u3.r2.b1, m1, NB, 0, B0, 256, 0, nullptr, 0, 256, 0, 1);
    // final conv: residual add, ReLU, mask, write f32 channel-major output
    C.conv(B0, 256, p_u3r2b, u3.r2.b2, m1, C0, 256, nullptr, 0, 0, fout,
           0, 256, 0, 1);
}
